// CrossAttentionFusion_49228915147527
// MI455X (gfx1250) — matrix operand order repaired and checked by reading, not compiled, not run on hardware
//
#include <hip/hip_runtime.h>
#include <hip/hip_bf16.h>

typedef __attribute__((ext_vector_type(16))) _Float16 v16h;
typedef __attribute__((ext_vector_type(8)))  _Float16 v8h;
typedef __attribute__((ext_vector_type(2)))  __fp16   fp16x2;
typedef __attribute__((ext_vector_type(8)))  float    v8f;
typedef __attribute__((ext_vector_type(4)))  unsigned u32x4;
typedef __attribute__((ext_vector_type(8)))  int      i32x8;
typedef __attribute__((ext_vector_type(4)))  int      i32x4;

static constexpr int NTOK = 4096;   // H*W
static constexpr int CFD  = 256;    // fused channels
static constexpr int NHD  = 8;      // heads
static constexpr int HD   = 32;     // head dim

union U16h { v16h h; unsigned u[8]; };
union U8h  { v8h  h; unsigned u[4]; };

__device__ __forceinline__ v8f wmma16(v16h a, v16h b, v8f c) {
  return __builtin_amdgcn_wmma_f32_16x16x32_f16(false, a, false, b, (short)0, c,
                                                false, false);
}

// v_cvt_pk_rtz_f16_f32: pack two floats into one dword of f16 pairs
__device__ __forceinline__ unsigned pkh(float a, float b) {
  union { fp16x2 h; unsigned u; } t;
  t.h = __builtin_amdgcn_cvt_pkrtz(a, b);
  return t.u;
}

// A-fragment (16x32 f16) from a row-major f16 row (global or LDS).
// e<8 -> k = (hi?8:0)+e ; e>=8 -> k = (hi?8:0)+16+(e-8)
__device__ __forceinline__ v16h load_a_h(const _Float16* rowp, int hi) {
  const int d0 = hi ? 8 : 0;
  v8h lo = *(const v8h*)(rowp + d0);
  v8h hh = *(const v8h*)(rowp + d0 + 16);
  v16h a;
#pragma unroll
  for (int i = 0; i < 8; ++i) { a[i] = lo[i]; a[i + 8] = hh[i]; }
  return a;
}

// ---------------------------------------------------------------------------
// TDM: async-copy one 32x32 f16 tile (row stride CFD halfs) global -> LDS.
// D# per CDNA5 ISA:
//  group0: [1:0]count=1 | [63:32]lds_addr | [120:64]global_addr | [127:126]type=2
//  group1: [17:16]data_size=1(2B) | tensor_dim0/1 large | tile 32x32 |
//          tensor_dim0_stride=CFD ; groups 2/3 unused (2D tensor)
// ---------------------------------------------------------------------------
__device__ __forceinline__ void tdm_load_tile(unsigned lds_addr,
                                              unsigned long long ga) {
  u32x4 g0;
  g0.x = 1u;                                     // count=1, user descriptor
  g0.y = lds_addr;                               // LDS byte address
  g0.z = (unsigned)ga;                           // global_addr[31:0]
  g0.w = (unsigned)((ga >> 32) & 0x01FFFFFFull)  // global_addr[56:32]
         | (2u << 30);                           // type=2 ("image")
  i32x8 g1;
  g1[0] = 0x00010000;            // wgmask=0, data_size=2B
  g1[1] = 0;                     // abar=0, tensor_dim0[15:0]=0 (dim0=1M)
  g1[2] = 16;                    // tensor_dim0[31:16]=16, tensor_dim1[15:0]=0
  g1[3] = (32 << 16) | 16;       // tensor_dim1[31:16]=16, tile_dim0=32
  g1[4] = 32;                    // tile_dim1=32, tile_dim2=0
  g1[5] = CFD;                   // tensor_dim0_stride[31:0]=256
  g1[6] = 0;                     // stride0 hi, stride1 lo
  g1[7] = 0;
  i32x4 z4 = {0, 0, 0, 0};
  i32x8 z8 = {0, 0, 0, 0, 0, 0, 0, 0};
  __builtin_amdgcn_tensor_load_to_lds(g0, g1, z4, z4, z8, 0);
}

// ---------------------------------------------------------------------------
// f32 -> f16 flat convert (weights)
// ---------------------------------------------------------------------------
__global__ __launch_bounds__(256) void cvt_kernel(const float* __restrict__ X,
                                                  _Float16* __restrict__ Y, int n) {
  const int i = blockIdx.x * 256 + threadIdx.x;
  if (i < n) Y[i] = (_Float16)X[i];
}

// ---------------------------------------------------------------------------
// NCHW feature map (C, NTOK) f32 -> token-major (NTOK, C) f16, via LDS tile
// ---------------------------------------------------------------------------
__global__ __launch_bounds__(256) void tcvt_kernel(const float* __restrict__ X,
                                                   _Float16* __restrict__ Y, int C) {
  __shared__ float tile[32][33];
  const int nb = blockIdx.x * 32;
  const int cb = blockIdx.y * 32;
  const int tx = threadIdx.x & 31;
  const int ty = threadIdx.x >> 5;
#pragma unroll
  for (int i = 0; i < 4; ++i) {
    const int c = cb + ty + i * 8;
    tile[ty + i * 8][tx] = X[(size_t)c * NTOK + nb + tx];
  }
  __syncthreads();
#pragma unroll
  for (int i = 0; i < 4; ++i) {
    const int n = nb + ty + i * 8;
    Y[(size_t)n * C + cb + tx] = (_Float16)tile[tx][ty + i * 8];
  }
}

// ---------------------------------------------------------------------------
// Unified GEMM: Y(token,m) = A(token,K) @ W(m,K)^T [+bias][relu]
// A, W both f16.  WG = 128 thr (4 waves): 64 tokens x 64 channels.
// ---------------------------------------------------------------------------
template <bool OUT_HALF, bool BIAS, bool RELU>
__global__ __launch_bounds__(128) void gemm_kernel(const _Float16* __restrict__ A,
                                                   const _Float16* __restrict__ W,
                                                   const float* __restrict__ bias,
                                                   int K, int M,
                                                   _Float16* __restrict__ Yh,
                                                   float* __restrict__ Yf) {
  const int nb   = blockIdx.x * 64;
  const int mb   = blockIdx.y * 64;
  const int w    = threadIdx.x >> 5;
  const int lane = threadIdx.x & 31;
  const int l16  = lane & 15;
  const int hi   = lane >> 4;
  const int tok  = nb + w * 16 + l16;

  v8f acc[4] = {};
  for (int kb = 0; kb < K; kb += 32) {
    v16h af = *(const v16h*)(A + (size_t)tok * K + kb + hi * 16);
#pragma unroll
    for (int j = 0; j < 4; ++j) {
      const int m = mb + j * 16 + l16;
      const _Float16* wp = W + (size_t)m * K + kb + (hi ? 16 : 0);
      __builtin_prefetch(wp + 32, 0, 1);                 // global_prefetch_b8
      v16h bf = *(const v16h*)wp;
      acc[j] = wmma16(af, bf, acc[j]);
    }
  }
#pragma unroll
  for (int j = 0; j < 4; ++j) {
    const int m = mb + j * 16 + l16;
    const float bv = BIAS ? bias[m] : 0.0f;
#pragma unroll
    for (int r = 0; r < 8; ++r) {
      const int n = nb + w * 16 + r + hi * 8;
      float v = acc[j][r] + bv;
      if (RELU) v = fmaxf(v, 0.0f);
      if (OUT_HALF) Yh[(size_t)n * M + m] = (_Float16)v;
      else          Yf[(size_t)n * M + m] = v;
    }
  }
}

// ---------------------------------------------------------------------------
// Flash attention, transposed-score scheme.
//   S^T = K_tile(16x32) x Q^T(32x16)  -> lane = query col, VGPRs = key rows
//   O^T = V^T(16x32) x P^T(32x16)     -> P^T built from registers via 8 shfl
// K tiles arrive via TDM (tensor_load_to_lds, wave 0), double-buffered;
// V tiles via register-prefetch + transposed LDS stores.
// ---------------------------------------------------------------------------
__global__ __launch_bounds__(128) void attn_kernel(const _Float16* __restrict__ Qh,
                                                   const _Float16* __restrict__ Kh,
                                                   const _Float16* __restrict__ Vh,
                                                   _Float16* __restrict__ Oh) {
  __shared__ alignas(32) _Float16 Kt[2][32][32];    // [buf][tok][d]  (TDM dest)
  __shared__ alignas(32) _Float16 VtT[2][32][32];   // [buf][d][tok]

  const int qb   = blockIdx.x * 64;
  const int h    = blockIdx.y;
  const int w    = threadIdx.x >> 5;
  const int lane = threadIdx.x & 31;
  const int l16  = lane & 15;
  const int hi   = lane >> 4;
  const float scale = 0.17677669529663687f;         // 1/sqrt(32)

  // Q^T B-fragment: lane = query col, e -> d = (hi?16:0)+e ; scale folded in
  const int q = qb + w * 16 + l16;
  v16h qB = *(const v16h*)(Qh + (size_t)q * CFD + h * HD + hi * 16);
#pragma unroll
  for (int e = 0; e < 16; ++e) qB[e] *= (_Float16)scale;

  const unsigned long long kbase =
      (unsigned long long)(const void*)(Kh + (size_t)h * HD);
  const unsigned klds0 = (unsigned)(size_t)&Kt[0][0][0];
  const unsigned klds1 = (unsigned)(size_t)&Kt[1][0][0];

  // cooperative V loader: 128 thr, tok = tid/4, 8-half chunk = tid%4
  const int tok = threadIdx.x >> 2;
  const int ch  = (threadIdx.x & 3) * 8;
  v8h vreg = *(const v8h*)(Vh + (size_t)tok * CFD + h * HD + ch);

  if (w == 0) tdm_load_tile(klds0, kbase);          // K tile for kb=0

  v8f O0 = {}, O1 = {};
  float mrun = -3.0e38f, lrun = 0.0f;

  for (int kb = 0; kb < NTOK; kb += 32) {
    const int cur = (kb >> 5) & 1;
#pragma unroll
    for (int i = 0; i < 8; ++i) VtT[cur][ch + i][tok] = vreg[i];
    if (kb + 32 < NTOK)
      vreg = *(const v8h*)(Vh + (size_t)(kb + 32 + tok) * CFD + h * HD + ch);
    if (w == 0) __builtin_amdgcn_s_wait_tensorcnt(0);   // K[cur] landed
    __syncthreads();
    if (w == 0 && kb + 32 < NTOK)                   // overlap next K with compute
      tdm_load_tile(cur ? klds0 : klds1,
                    kbase + (unsigned long long)(kb + 32) * CFD * 2ull);

    // scores^T: two 16-key tiles
    v16h kA0 = *(const v16h*)&Kt[cur][l16][hi * 16];
    v16h kA1 = *(const v16h*)&Kt[cur][16 + l16][hi * 16];
    v8f S0 = {}, S1 = {};
    S0 = wmma16(kA0, qB, S0);
    S1 = wmma16(kA1, qB, S1);

    // online softmax over this lane's 16 keys + partner's 16 keys
    float mloc = S0[0];
#pragma unroll
    for (int r = 1; r < 8; ++r) mloc = fmaxf(mloc, S0[r]);
#pragma unroll
    for (int r = 0; r < 8; ++r) mloc = fmaxf(mloc, S1[r]);
    mloc = fmaxf(mloc, __shfl_xor(mloc, 16));
    const float mnew  = fmaxf(mrun, mloc);
    const float alpha = __expf(mrun - mnew);
    float p0[8], p1[8], sloc = 0.0f;
#pragma unroll
    for (int r = 0; r < 8; ++r) { p0[r] = __expf(S0[r] - mnew); sloc += p0[r]; }
#pragma unroll
    for (int r = 0; r < 8; ++r) { p1[r] = __expf(S1[r] - mnew); sloc += p1[r]; }
    sloc += __shfl_xor(sloc, 16);
    lrun = lrun * alpha + sloc;
    mrun = mnew;
#pragma unroll
    for (int r = 0; r < 8; ++r) { O0[r] *= alpha; O1[r] *= alpha; }

    // build P^T B-fragment: lane (l16,hi) needs keys (hi?16:0)+0..15 for query l16
    unsigned own0[4], own1[4], oth0[4], oth1[4];
#pragma unroll
    for (int i = 0; i < 4; ++i) {
      own0[i] = pkh(p0[2 * i], p0[2 * i + 1]);
      own1[i] = pkh(p1[2 * i], p1[2 * i + 1]);
    }
#pragma unroll
    for (int i = 0; i < 4; ++i) {
      oth0[i] = __shfl_xor(own0[i], 16);
      oth1[i] = __shfl_xor(own1[i], 16);
    }
    U16h pB;
#pragma unroll
    for (int i = 0; i < 4; ++i) {
      pB.u[i]     = hi ? oth1[i] : own0[i];   // keys (hi?16:0)+0..7
      pB.u[4 + i] = hi ? own1[i] : oth0[i];   // keys (hi?16:0)+8..15
    }

    // O^T += V^T x P^T  (two d-row tiles)
    v16h vA0 = *(const v16h*)&VtT[cur][l16][hi * 16];
    v16h vA1 = *(const v16h*)&VtT[cur][16 + l16][hi * 16];
    O0 = wmma16(vA0, pB.h, O0);
    O1 = wmma16(vA1, pB.h, O1);
  }

  // finalize: lane = query col, VGPR r -> d = r + 8*hi (+16 for tile 1)
  const float inv = 1.0f / lrun;
  U8h o0, o1;
#pragma unroll
  for (int i = 0; i < 4; ++i) {
    o0.u[i] = pkh(O0[2 * i] * inv, O0[2 * i + 1] * inv);
    o1.u[i] = pkh(O1[2 * i] * inv, O1[2 * i + 1] * inv);
  }
  _Float16* obase = Oh + (size_t)q * CFD + h * HD;
  *(v8h*)(obase + 8 * hi)      = o0.h;
  *(v8h*)(obase + 16 + 8 * hi) = o1.h;
}

// ---------------------------------------------------------------------------
// LayerNorm over CF=256 channels, one block per token (256 thr = 8 waves).
// ---------------------------------------------------------------------------
__device__ __forceinline__ float ln_block_reduce(float v, float* sh, int tid) {
  float s = v;
#pragma unroll
  for (int off = 16; off >= 1; off >>= 1) s += __shfl_xor(s, off);
  if ((tid & 31) == 0) sh[tid >> 5] = s;
  __syncthreads();
  float a = (tid < 8) ? sh[tid] : 0.0f;
  if (tid < 8) {
#pragma unroll
    for (int off = 4; off >= 1; off >>= 1) a += __shfl_xor(a, off);
    if (tid == 0) sh[0] = a;
  }
  __syncthreads();
  const float r = sh[0];
  __syncthreads();
  return r;
}

__global__ __launch_bounds__(256) void ln1_kernel(const float* __restrict__ resf,
                                                  const float* __restrict__ projf,
                                                  const float* __restrict__ g,
                                                  const float* __restrict__ b,
                                                  _Float16* __restrict__ xh,
                                                  float* __restrict__ xf) {
  __shared__ float sh[8];
  const int n = blockIdx.x, c = threadIdx.x;
  const float v = resf[(size_t)n * CFD + c] + projf[(size_t)n * CFD + c];
  const float s1 = ln_block_reduce(v, sh, c);
  const float s2 = ln_block_reduce(v * v, sh, c);
  const float mean = s1 * (1.0f / CFD);
  const float var  = s2 * (1.0f / CFD) - mean * mean;
  const float y = (v - mean) * rsqrtf(var + 1e-5f) * g[c] + b[c];
  xh[(size_t)n * CFD + c] = (_Float16)y;
  xf[(size_t)n * CFD + c] = y;
}

__global__ __launch_bounds__(256) void ln2_kernel(const float* __restrict__ xf,
                                                  const float* __restrict__ ffnf,
                                                  const float* __restrict__ g,
                                                  const float* __restrict__ b,
                                                  float* __restrict__ out) {
  __shared__ float sh[8];
  const int n = blockIdx.x, c = threadIdx.x;
  const float v = xf[(size_t)n * CFD + c] + ffnf[(size_t)n * CFD + c];
  const float s1 = ln_block_reduce(v, sh, c);
  const float s2 = ln_block_reduce(v * v, sh, c);
  const float mean = s1 * (1.0f / CFD);
  const float var  = s2 * (1.0f / CFD) - mean * mean;
  const float y = (v - mean) * rsqrtf(var + 1e-5f) * g[c] + b[c];
  out[(size_t)c * NTOK + n] = y;                   // (B, CF, H, W) transpose
}

// ---------------------------------------------------------------------------
extern "C" void kernel_launch(void* const* d_in, const int* in_sizes, int n_in,
                              void* d_out, int out_size, void* d_ws, size_t ws_size,
                              hipStream_t stream) {
  (void)in_sizes; (void)n_in; (void)out_size; (void)ws_size;
  const float* F_lidar = (const float*)d_in[0];
  const float* F_cam   = (const float*)d_in[1];
  const float* Wq      = (const float*)d_in[2];
  const float* Wk      = (const float*)d_in[3];
  const float* Wv      = (const float*)d_in[4];
  const float* Wo      = (const float*)d_in[5];
  const float* Wres    = (const float*)d_in[6];
  const float* ln1_g   = (const float*)d_in[7];
  const float* ln1_b   = (const float*)d_in[8];
  const float* ln2_g   = (const float*)d_in[9];
  const float* ln2_b   = (const float*)d_in[10];
  const float* W1      = (const float*)d_in[11];
  const float* b1      = (const float*)d_in[12];
  const float* W2      = (const float*)d_in[13];
  const float* b2      = (const float*)d_in[14];
  float* out = (float*)d_out;

  char* ws = (char*)d_ws;
  size_t off = 0;
  auto carve = [&](size_t bytes) -> void* {
    void* p = ws + off;
    off += (bytes + 255) & ~(size_t)255;
    return p;
  };
  _Float16* Lh    = (_Float16*)carve((size_t)NTOK * 64 * 2);    // tok_l f16
  _Float16* Ch    = (_Float16*)carve((size_t)NTOK * CFD * 2);   // tok_c f16
  _Float16* Qh    = (_Float16*)carve((size_t)NTOK * CFD * 2);
  _Float16* Kh    = (_Float16*)carve((size_t)NTOK * CFD * 2);
  _Float16* Vh    = (_Float16*)carve((size_t)NTOK * CFD * 2);
  _Float16* AttnH = (_Float16*)carve((size_t)NTOK * CFD * 2);
  _Float16* Xh    = (_Float16*)carve((size_t)NTOK * CFD * 2);
  _Float16* Hh    = (_Float16*)carve((size_t)NTOK * 4 * CFD * 2);
  _Float16* Wqh   = (_Float16*)carve((size_t)CFD * 64 * 2);
  _Float16* Wkh   = (_Float16*)carve((size_t)CFD * CFD * 2);
  _Float16* Wvh   = (_Float16*)carve((size_t)CFD * CFD * 2);
  _Float16* Woh   = (_Float16*)carve((size_t)CFD * CFD * 2);
  _Float16* Wrh   = (_Float16*)carve((size_t)CFD * 64 * 2);
  _Float16* W1h   = (_Float16*)carve((size_t)4 * CFD * CFD * 2);
  _Float16* W2h   = (_Float16*)carve((size_t)4 * CFD * CFD * 2);
  float*    Resf  = (float*)carve((size_t)NTOK * CFD * 4);
  float*    Projf = (float*)carve((size_t)NTOK * CFD * 4);
  float*    Xf    = (float*)carve((size_t)NTOK * CFD * 4);
  float*    Ffnf  = (float*)carve((size_t)NTOK * CFD * 4);

  const dim3 blk(128);
  auto cvt = [&](const float* src, _Float16* dst, int n) {
    cvt_kernel<<<(n + 255) / 256, 256, 0, stream>>>(src, dst, n);
  };
  // weights -> f16
  cvt(Wq, Wqh, CFD * 64);
  cvt(Wk, Wkh, CFD * CFD);
  cvt(Wv, Wvh, CFD * CFD);
  cvt(Wo, Woh, CFD * CFD);
  cvt(Wres, Wrh, CFD * 64);
  cvt(W1, W1h, 4 * CFD * CFD);
  cvt(W2, W2h, 4 * CFD * CFD);
  // feature maps -> token-major f16
  tcvt_kernel<<<dim3(NTOK / 32, 64 / 32), 256, 0, stream>>>(F_lidar, Lh, 64);
  tcvt_kernel<<<dim3(NTOK / 32, CFD / 32), 256, 0, stream>>>(F_cam, Ch, CFD);

  // projections (1x1 convs)
  gemm_kernel<true,  false, false><<<dim3(NTOK / 64, CFD / 64), blk, 0, stream>>>(
      Lh, Wqh, nullptr, 64, CFD, Qh, nullptr);
  gemm_kernel<true,  false, false><<<dim3(NTOK / 64, CFD / 64), blk, 0, stream>>>(
      Ch, Wkh, nullptr, CFD, CFD, Kh, nullptr);
  gemm_kernel<true,  false, false><<<dim3(NTOK / 64, CFD / 64), blk, 0, stream>>>(
      Ch, Wvh, nullptr, CFD, CFD, Vh, nullptr);
  gemm_kernel<false, false, false><<<dim3(NTOK / 64, CFD / 64), blk, 0, stream>>>(
      Lh, Wrh, nullptr, 64, CFD, nullptr, Resf);
  // flash attention (TDM-fed K tiles)
  attn_kernel<<<dim3(NTOK / 64, NHD), blk, 0, stream>>>(Qh, Kh, Vh, AttnH);
  // out projection
  gemm_kernel<false, false, false><<<dim3(NTOK / 64, CFD / 64), blk, 0, stream>>>(
      AttnH, Woh, nullptr, CFD, CFD, nullptr, Projf);
  // LN1(res + out)
  ln1_kernel<<<NTOK, 256, 0, stream>>>(Resf, Projf, ln1_g, ln1_b, Xh, Xf);
  // FFN
  gemm_kernel<true, true, true><<<dim3(NTOK / 64, (4 * CFD) / 64), blk, 0, stream>>>(
      Xh, W1h, b1, CFD, 4 * CFD, Hh, nullptr);
  gemm_kernel<false, true, false><<<dim3(NTOK / 64, CFD / 64), blk, 0, stream>>>(
      Hh, W2h, b2, 4 * CFD, CFD, nullptr, Ffnf);
  // LN2(x + ffn) + transpose to (B, CF, H, W)
  ln2_kernel<<<NTOK, 256, 0, stream>>>(Xf, Ffnf, ln2_g, ln2_b, out);
}
